// QuantizedAdaptiveRankLinear_1391569404121
// MI455X (gfx1250) — 8-bit matrix operand order repaired and checked by reading, compiled, not run on hardware
//
#include <hip/hip_runtime.h>


typedef __attribute__((ext_vector_type(8))) int v8i;

#define WS_ABSX  0
#define WS_IMAX  4
#define WS_ABSG  16
#define WS_G0    64
#define WS_G3    1024
#define WS_G1    4096
#define WS_G2    36864

constexpr int BT = 4;
constexpr int X0 = 256 * BT;
constexpr int X1 = 128 * BT;
constexpr int X2 = 128 * BT;
constexpr int X3 = 256 * BT;

__device__ __forceinline__ v8i wmma_iu8(v8i a, v8i b, v8i c) {
  return __builtin_amdgcn_wmma_i32_16x16x64_iu8(true, a, true, b, c, false, false);
}

__device__ __forceinline__ int a_koff(int v, int g) {
  return ((v >> 2) << 5) + (g << 4) + ((v & 3) << 2);
}
__device__ __forceinline__ int b_koff(int v, int g) {
  return ((v >> 2) << 5) + (g << 4) + ((v & 3) << 2);
}

__device__ __forceinline__ v8i load_afrag192(const unsigned char* p, int row,
                                             int kstep, int g) {
  v8i a;
#pragma unroll
  for (int v = 0; v < 8; ++v)
    a[v] = *(const int*)(p + row * 192 + kstep + a_koff(v, g));
  return a;
}

__device__ __forceinline__ v8i load_bfrag192(const unsigned char* p, int col,
                                             int kstep, int g) {
  v8i b;
#pragma unroll
  for (int v = 0; v < 8; ++v)
    b[v] = *(const int*)(p + col * 192 + kstep + b_koff(v, g));
  return b;
}

__global__ void init_ws_kernel(unsigned char* ws) {
  if (threadIdx.x < 4) ((unsigned int*)ws)[threadIdx.x] = 0u;
}

__global__ __launch_bounds__(256) void absmax_x_kernel(const float* __restrict__ x,
                                                       int n, unsigned char* ws) {
  __shared__ unsigned int sm;
  if (threadIdx.x == 0) sm = 0u;
  __syncthreads();
  unsigned int lm = 0u;
  for (int i = blockIdx.x * blockDim.x + threadIdx.x; i < n;
       i += gridDim.x * blockDim.x)
    lm = max(lm, __float_as_uint(fabsf(x[i])));
  atomicMax(&sm, lm);
  __syncthreads();
  if (threadIdx.x == 0) atomicMax((unsigned int*)(ws + WS_ABSX), sm);
}

__global__ __launch_bounds__(256) void pack_cores_kernel(
    const float* __restrict__ c0, const float* __restrict__ c1,
    const float* __restrict__ c2, const float* __restrict__ c3,
    unsigned char* ws) {
  __shared__ unsigned int sm;
  __shared__ float sAbs[4];
  const int tid = threadIdx.x;
  const float* cores[4] = {c0, c1, c2, c3};
  const int sizes[4] = {320, 25600, 25600, 320};
  float* absg = (float*)(ws + WS_ABSG);

  for (int k = 0; k < 4; ++k) {
    if (tid == 0) sm = 0u;
    __syncthreads();
    unsigned int lm = 0u;
    for (int i = tid; i < sizes[k]; i += 256)
      lm = max(lm, __float_as_uint(fabsf(cores[k][i])));
    atomicMax(&sm, lm);
    __syncthreads();
    if (tid == 0) {
      sAbs[k] = __uint_as_float(sm);
      absg[k] = sAbs[k];
    }
    __syncthreads();
  }
  float s[4];
#pragma unroll
  for (int k = 0; k < 4; ++k) s[k] = fmaxf(sAbs[k] / 127.0f, 1e-8f);

  unsigned int* g0 = (unsigned int*)(ws + WS_G0);
  for (int o = tid; o < 80; o += 256) {
    int r = o >> 2, n = o & 3;
    unsigned int w = 0u;
#pragma unroll
    for (int m = 0; m < 4; ++m) {
      int q = __float2int_rn(c0[(m * 4 + n) * 20 + r] / s[0]);
      w |= (unsigned int)(q & 255) << (8 * m);
    }
    g0[o] = w;
  }
  unsigned char* g1 = ws + WS_G1;
  for (int idx = tid; idx < 160 * 192; idx += 256) {
    int o = idx / 192, kk = idx % 192;
    int q = 0;
    if (kk < 160) {
      int r2 = o >> 3, n2 = o & 7, r1 = kk >> 3, m2 = kk & 7;
      q = __float2int_rn(c1[((r1 * 8 + m2) * 8 + n2) * 20 + r2] / s[1]);
    }
    g1[idx] = (unsigned char)(q & 255);
  }
  unsigned char* g2 = ws + WS_G2;
  for (int idx = tid; idx < 160 * 192; idx += 256) {
    int o = idx / 192, kk = idx % 192;
    int q = 0;
    if (kk < 160) {
      int r3 = o >> 3, n3 = o & 7, r2 = kk >> 3, m3 = kk & 7;
      q = __float2int_rn(c2[((r2 * 8 + m3) * 8 + n3) * 20 + r3] / s[2]);
    }
    g2[idx] = (unsigned char)(q & 255);
  }
  unsigned char* g3 = ws + WS_G3;
  for (int idx = tid; idx < 16 * 80; idx += 256) {
    int o = idx / 80, kk = idx % 80;
    int q = 0;
    if (o < 4) {
      int r3 = kk >> 2, m4 = kk & 3;
      q = __float2int_rn(c3[(r3 * 4 + m4) * 4 + o] / s[3]);
    }
    g3[idx] = (unsigned char)(q & 255);
  }
}

template <int PASS>
__global__ __launch_bounds__(256) void tt_pass_kernel(
    const float* __restrict__ x, const float* __restrict__ bias,
    float* __restrict__ out, unsigned char* ws) {
  __shared__ unsigned char sT1[X1 * 192];
  __shared__ unsigned char sT2[X2 * 192];
  __shared__ unsigned int sT0[X0];
  __shared__ int sMax;

  const int tid = threadIdx.x;
  const int lane = tid & 31;
  const int wave = tid >> 5;
  const int g = lane >> 4;
  const int ln = lane & 15;
  const int b0 = blockIdx.x * BT;

  const float absx = __uint_as_float(*(const unsigned int*)(ws + WS_ABSX));
  const float* absg = (const float*)(ws + WS_ABSG);
  const int* imax = (const int*)(ws + WS_IMAX);
  const float sx = fmaxf(absx / 127.0f, 1e-8f);
  const float sg0 = fmaxf(absg[0] / 127.0f, 1e-8f);
  const float sg1 = fmaxf(absg[1] / 127.0f, 1e-8f);
  const float sg2 = fmaxf(absg[2] / 127.0f, 1e-8f);
  const float sg3 = fmaxf(absg[3] / 127.0f, 1e-8f);
  float f0 = 0.f, f1 = 0.f, f2 = 0.f, fout = 0.f;
  float sp = sx * sg0;
  if constexpr (PASS > 0) {
    float s1 = fmaxf(sp * (float)imax[0] / 127.0f, 1e-8f);
    f0 = sp / s1;
    sp = s1 * sg1;
  }
  if constexpr (PASS > 1) {
    float s2 = fmaxf(sp * (float)imax[1] / 127.0f, 1e-8f);
    f1 = sp / s2;
    sp = s2 * sg2;
  }
  if constexpr (PASS > 2) {
    float s3 = fmaxf(sp * (float)imax[2] / 127.0f, 1e-8f);
    f2 = sp / s3;
    fout = s3 * sg3;
  }

  if (tid == 0) sMax = 0;

  __builtin_prefetch(ws + WS_G1 + ((tid & 255) << 7), 0, 3);
  __builtin_prefetch(ws + WS_G2 + ((tid & 255) << 7), 0, 3);

  if constexpr (PASS > 0)
    for (int i = tid; i < X1 * 192 / 4; i += 256) ((unsigned int*)sT1)[i] = 0u;
  if constexpr (PASS > 1)
    for (int i = tid; i < X2 * 192 / 4; i += 256) ((unsigned int*)sT2)[i] = 0u;

  for (int x0 = tid; x0 < X0; x0 += 256) {
    int fg = x0 / BT, b = x0 % BT;
    const float* xr = x + (size_t)(b0 + b) * 1024 + fg;
    unsigned int w = 0u;
#pragma unroll
    for (int m = 0; m < 4; ++m) {
      int q = __float2int_rn(xr[m * 256] / sx);
      w |= (unsigned int)(q & 255) << (8 * m);
    }
    sT0[x0] = w;
  }
  __syncthreads();

  int lmax = 0;

  {
    const unsigned int* g0p = (const unsigned int*)(ws + WS_G0);
    for (int nt = 0; nt < 5; ++nt) {
      v8i a = {0, 0, 0, 0, 0, 0, 0, 0};
      if (g == 0) a[0] = (int)g0p[nt * 16 + ln];
      for (int xt = wave; xt < X0 / 16; xt += 8) {
        v8i bf = {0, 0, 0, 0, 0, 0, 0, 0};
        if (g == 0) bf[0] = (int)sT0[xt * 16 + ln];
        v8i c = {0, 0, 0, 0, 0, 0, 0, 0};
        c = wmma_iu8(a, bf, c);
        const int xx = xt * 16 + ln;
        if constexpr (PASS == 0) {
#pragma unroll
          for (int vj = 0; vj < 8; ++vj) {
            int acc = c[vj];
            lmax = max(lmax, acc < 0 ? -acc : acc);
          }
        } else {
          const int mp = xx / (X0 / 8), xr2 = xx % (X0 / 8);
#pragma unroll
          for (int vj = 0; vj < 8; ++vj) {
            int o = nt * 16 + vj + 8 * g;
            int q = __float2int_rn((float)c[vj] * f0);
            int r = o >> 2, n = o & 3;
            sT1[(xr2 * 4 + n) * 192 + r * 8 + mp] = (unsigned char)(q & 255);
          }
        }
      }
    }
  }
  __syncthreads();

  if constexpr (PASS >= 1) {
    const unsigned char* g1p = ws + WS_G1;
    for (int nt = 0; nt < 10; ++nt) {
      v8i a0f = load_afrag192(g1p, nt * 16 + ln, 0, g);
      v8i a1f = load_afrag192(g1p, nt * 16 + ln, 64, g);
      v8i a2f = load_afrag192(g1p, nt * 16 + ln, 128, g);
      for (int xt = wave; xt < X1 / 16; xt += 8) {
        int col = xt * 16 + ln;
        v8i c = {0, 0, 0, 0, 0, 0, 0, 0};
        c = wmma_iu8(a0f, load_bfrag192(sT1, col, 0, g), c);
        c = wmma_iu8(a1f, load_bfrag192(sT1, col, 64, g), c);
        c = wmma_iu8(a2f, load_bfrag192(sT1, col, 128, g), c);
        if constexpr (PASS == 1) {
#pragma unroll
          for (int vj = 0; vj < 8; ++vj) {
            int acc = c[vj];
            lmax = max(lmax, acc < 0 ? -acc : acc);
          }
        } else {
          const int mp = col / (X1 / 8), xr2 = col % (X1 / 8);
#pragma unroll
          for (int vj = 0; vj < 8; ++vj) {
            int o = nt * 16 + vj + 8 * g;
            int q = __float2int_rn((float)c[vj] * f1);
            int r = o >> 3, n = o & 7;
            sT2[(xr2 * 8 + n) * 192 + r * 8 + mp] = (unsigned char)(q & 255);
          }
        }
      }
    }
  }
  __syncthreads();

  if constexpr (PASS >= 2) {
    const unsigned char* g2p = ws + WS_G2;
    for (int nt = 0; nt < 10; ++nt) {
      v8i a0f = load_afrag192(g2p, nt * 16 + ln, 0, g);
      v8i a1f = load_afrag192(g2p, nt * 16 + ln, 64, g);
      v8i a2f = load_afrag192(g2p, nt * 16 + ln, 128, g);
      for (int xt = wave; xt < X2 / 16; xt += 8) {
        int col = xt * 16 + ln;
        v8i c = {0, 0, 0, 0, 0, 0, 0, 0};
        c = wmma_iu8(a0f, load_bfrag192(sT2, col, 0, g), c);
        c = wmma_iu8(a1f, load_bfrag192(sT2, col, 64, g), c);
        c = wmma_iu8(a2f, load_bfrag192(sT2, col, 128, g), c);
        if constexpr (PASS == 2) {
#pragma unroll
          for (int vj = 0; vj < 8; ++vj) {
            int acc = c[vj];
            lmax = max(lmax, acc < 0 ? -acc : acc);
          }
        } else {
          const int mp = col / (X2 / 4), xr2 = col % (X2 / 4);
#pragma unroll
          for (int vj = 0; vj < 8; ++vj) {
            int o = nt * 16 + vj + 8 * g;
            int q = __float2int_rn((float)c[vj] * f2);
            int r = o >> 3, n = o & 7;
            sT1[(xr2 * 8 + n) * 80 + r * 4 + mp] = (unsigned char)(q & 255);
          }
        }
      }
    }
  }
  __syncthreads();

  if constexpr (PASS == 3) {
    const unsigned char* g3p = ws + WS_G3;
    v8i a0f, a1f;
#pragma unroll
    for (int v = 0; v < 8; ++v) {
      a0f[v] = *(const int*)(g3p + ln * 80 + a_koff(v, g));
      a1f[v] = (v < 4 && g == 0) ? *(const int*)(g3p + ln * 80 + 64 + a_koff(v, g)) : 0;
    }
    for (int xt = wave; xt < X3 / 16; xt += 8) {
      int col = xt * 16 + ln;
      v8i b0f, b1f;
#pragma unroll
      for (int v = 0; v < 8; ++v) {
        b0f[v] = *(const int*)(sT1 + col * 80 + b_koff(v, g));
        b1f[v] = (v < 4 && g == 0)
                     ? *(const int*)(sT1 + col * 80 + 64 + b_koff(v, g))
                     : 0;
      }
      v8i c = {0, 0, 0, 0, 0, 0, 0, 0};
      c = wmma_iu8(a0f, b0f, c);
      c = wmma_iu8(a1f, b1f, c);
      const int brow = b0 + (col >> 8);
      const int ocolb = (col & 255) * 4;
#pragma unroll
      for (int vj = 0; vj < 4; ++vj) {
        if (g == 0) {
          out[(size_t)brow * 1024 + ocolb + vj] =
              fout * (float)c[vj] + bias[ocolb + vj];
        }
      }
    }
  }

  if constexpr (PASS < 3) {
    atomicMax(&sMax, lmax);
    __syncthreads();
    if (tid == 0) atomicMax((int*)(ws + WS_IMAX) + PASS, sMax);
  }
}

extern "C" void kernel_launch(void* const* d_in, const int* in_sizes, int n_in,
                              void* d_out, int out_size, void* d_ws,
                              size_t ws_size, hipStream_t stream) {
  const float* x = (const float*)d_in[0];
  const float* c0 = (const float*)d_in[1];
  const float* c1 = (const float*)d_in[2];
  const float* c2 = (const float*)d_in[3];
  const float* c3 = (const float*)d_in[4];
  const float* bias = (const float*)d_in[5];
  float* out = (float*)d_out;
  unsigned char* ws = (unsigned char*)d_ws;

  const int nx = in_sizes[0];
  const int B = nx / 1024;
  const int nblocks = (B + BT - 1) / BT;

  init_ws_kernel<<<1, 64, 0, stream>>>(ws);
  absmax_x_kernel<<<512, 256, 0, stream>>>(x, nx, ws);
  pack_cores_kernel<<<1, 256, 0, stream>>>(c0, c1, c2, c3, ws);
  tt_pass_kernel<0><<<nblocks, 256, 0, stream>>>(x, bias, out, ws);
  tt_pass_kernel<1><<<nblocks, 256, 0, stream>>>(x, bias, out, ws);
  tt_pass_kernel<2><<<nblocks, 256, 0, stream>>>(x, bias, out, ws);
  tt_pass_kernel<3><<<nblocks, 256, 0, stream>>>(x, bias, out, ws);
}
